// _RHNCoupled_37520834298157
// MI455X (gfx1250) — matrix operand order repaired and checked by reading, not compiled, not run on hardware
//
#include <hip/hip_runtime.h>
#include <hip/hip_bf16.h>

typedef __attribute__((ext_vector_type(16))) __bf16 v16bf;
typedef __attribute__((ext_vector_type(4)))  __bf16 v4bf;
typedef __attribute__((ext_vector_type(8)))  float  v8f;

#define SEQ    200
#define BATCH  128
#define HIDDIM 1024
#define DEPTH  5

#define LDS_STRIDE 36  // 32 data bf16 + 4 pad -> 72B rows: 8B aligned, bank-conflict friendly

// ---------------- fragment helpers (wave32, v_wmma_f32_16x16x32_bf16) ----------------

// B matrix (K x N) fragment. Weights stored row-major (N, K) => element (k,n) = W[n*K + k].
// Layout: lanes 0-15 -> K 0..15, lanes 16-31 -> K 16..31; column n = lane & 15;
// the 16 bf16 per lane are consecutive K -> one contiguous 32B read per lane (L2-resident).
__device__ inline v16bf load_b_frag(const __bf16* __restrict__ W, int n0, int k0, int lane) {
  int n  = n0 + (lane & 15);
  int kh = lane >> 4;
  return *(const v16bf*)(W + (size_t)n * HIDDIM + k0 + kh * 16);
}

// A matrix (16 x 32) fragment from LDS.
// Contiguous k order, same as load_b_frag: m = lane & 15; element j -> k = j + (lane >> 4)*16.
__device__ inline v16bf load_a_frag(const __bf16* __restrict__ ldsA, int m0, int lane) {
  const __bf16* row = ldsA + (m0 + (lane & 15)) * LDS_STRIDE + (lane >> 4) * 16;
  v4bf g0 = *(const v4bf*)(row + 0);   // K base + 0..3
  v4bf g1 = *(const v4bf*)(row + 4);   // K base + 4..7
  v4bf g2 = *(const v4bf*)(row + 8);   // K base + 8..11
  v4bf g3 = *(const v4bf*)(row + 12);  // K base + 12..15
  v16bf a;
#pragma unroll
  for (int i = 0; i < 4; ++i) {
    a[i]      = g0[i];
    a[4 + i]  = g1[i];
    a[8 + i]  = g2[i];
    a[12 + i] = g3[i];
  }
  return a;
}

__device__ inline v8f wmma_bf16(v16bf a, v16bf b, v8f c) {
  return __builtin_amdgcn_wmma_f32_16x16x32_bf16(false, a, false, b, (short)0, c,
                                                 false, false);
}

// ---------------- small utility kernels ----------------

__global__ void cvt_f32_bf16(const float* __restrict__ src, __bf16* __restrict__ dst, int n) {
  int i = blockIdx.x * blockDim.x + threadIdx.x;
  if (i < n) dst[i] = (__bf16)src[i];
}

__global__ void copy_f32(const float* __restrict__ src, float* __restrict__ dst, int n) {
  int i = blockIdx.x * blockDim.x + threadIdx.x;
  if (i < n) dst[i] = src[i];
}

// ---------------- input projection: H/T = (inp * h_mask) @ W^T + b ----------------
// Grid: x = gate*8 + n_tile (16), y = m_tile (200). Block 256 = 8 waves (2 M x 4 N).
// Per wave: 64(M) x 32(N) = 4x2 WMMA tiles, fp32 accumulate.
__global__ void __launch_bounds__(256) rhn_input_gemm(
    const float* __restrict__ inp, const float* __restrict__ hmask,
    const __bf16* __restrict__ Wih, const __bf16* __restrict__ Wit,
    const float* __restrict__ bih, const float* __restrict__ bit,
    float* __restrict__ Hbuf, float* __restrict__ Tbuf)
{
  __shared__ __bf16 ldsA[128 * LDS_STRIDE];
  const int tid  = threadIdx.x;
  const int lane = tid & 31;
  const int wave = tid >> 5;
  const int gate = blockIdx.x >> 3;
  const int n0   = (blockIdx.x & 7) * 128;
  const int m0   = blockIdx.y * 128;

  const __bf16* W    = gate ? Wit : Wih;
  const float*  bias = gate ? bit : bih;
  float*        out  = gate ? Tbuf : Hbuf;

  const int wm  = wave & 1;          // M wave group (0/1)
  const int wn  = wave >> 1;         // N wave group (0..3)
  const int wMb = wm * 64;
  const int wNb = n0 + wn * 32;

  v8f zero = {0.f, 0.f, 0.f, 0.f, 0.f, 0.f, 0.f, 0.f};
  v8f acc[4][2];
#pragma unroll
  for (int i = 0; i < 4; ++i)
#pragma unroll
    for (int j = 0; j < 2; ++j) acc[i][j] = zero;

  // cooperative A staging: 256 threads cover 128 rows x 2 segments of 16 f32
  const int arow = tid >> 1;
  const int aseg = (tid & 1) * 16;
  const float* xrow = inp   + (size_t)(m0 + arow) * HIDDIM;
  const float* mrow = hmask + (size_t)((m0 + arow) & (BATCH - 1)) * HIDDIM;

  for (int k0 = 0; k0 < HIDDIM; k0 += 32) {
    {
      const float4* xs = (const float4*)(xrow + k0 + aseg);
      const float4* ms = (const float4*)(mrow + k0 + aseg);
      __bf16* dst = ldsA + arow * LDS_STRIDE + aseg;
#pragma unroll
      for (int q = 0; q < 4; ++q) {
        float4 xv = xs[q], mv = ms[q];
        v4bf o;
        o[0] = (__bf16)(xv.x * mv.x);
        o[1] = (__bf16)(xv.y * mv.y);
        o[2] = (__bf16)(xv.z * mv.z);
        o[3] = (__bf16)(xv.w * mv.w);
        *(v4bf*)(dst + q * 4) = o;
      }
    }
    __syncthreads();

    if (k0 + 32 < HIDDIM)
      __builtin_prefetch(W + (size_t)(wNb + (lane & 15)) * HIDDIM + k0 + 32, 0, 1);

    v16bf bfr[2];
#pragma unroll
    for (int j = 0; j < 2; ++j) bfr[j] = load_b_frag(W, wNb + j * 16, k0, lane);
#pragma unroll
    for (int i = 0; i < 4; ++i) {
      v16bf af = load_a_frag(ldsA, wMb + i * 16, lane);
#pragma unroll
      for (int j = 0; j < 2; ++j) acc[i][j] = wmma_bf16(af, bfr[j], acc[i][j]);
    }
    __syncthreads();
  }

  // D layout: m = r + (lane>>4)*8, n = lane & 15
  const int nl = lane & 15;
  const int mh = (lane >> 4) * 8;
#pragma unroll
  for (int j = 0; j < 2; ++j) {
    int   n  = wNb + j * 16 + nl;
    float bv = bias[n];
#pragma unroll
    for (int i = 0; i < 4; ++i) {
      float* orow = out + (size_t)(m0 + wMb + i * 16 + mh) * HIDDIM + n;
#pragma unroll
      for (int r = 0; r < 8; ++r) orow[(size_t)r * HIDDIM] = acc[i][j][r] + bv;
    }
  }
}

// ---------------- one recurrent highway layer (both gates fused) ----------------
// s_new = (tanh(pre_h) - s) * sigmoid(pre_t) + s
// Grid: 32 blocks (N tiles of 32). Block 128 = 4 waves (2 M x 2 N); wave = 64(M) x 16(N).
__global__ void __launch_bounds__(128) rhn_layer(
    const float* __restrict__ s_in, float* __restrict__ s_out,
    const float* __restrict__ smask,
    const __bf16* __restrict__ Wh, const __bf16* __restrict__ Wt,
    const float* __restrict__ bh, const float* __restrict__ bt,
    const float* __restrict__ Hs, const float* __restrict__ Ts,
    float* __restrict__ outs)
{
  __shared__ __bf16 ldsA[128 * LDS_STRIDE];
  const int tid  = threadIdx.x;
  const int lane = tid & 31;
  const int wave = tid >> 5;
  const int n0   = blockIdx.x * 32;
  const int wm   = wave & 1;
  const int wn   = wave >> 1;
  const int wMb  = wm * 64;
  const int wNb  = n0 + wn * 16;

  v8f zero = {0.f, 0.f, 0.f, 0.f, 0.f, 0.f, 0.f, 0.f};
  v8f accH[4], accT[4];
#pragma unroll
  for (int i = 0; i < 4; ++i) { accH[i] = zero; accT[i] = zero; }

  // staging: 128 threads, one full 32-element row segment each (masked state -> bf16)
  const float* srow = s_in  + (size_t)tid * HIDDIM;
  const float* mrow = smask + (size_t)tid * HIDDIM;

  for (int k0 = 0; k0 < HIDDIM; k0 += 32) {
    {
      const float4* xs = (const float4*)(srow + k0);
      const float4* ms = (const float4*)(mrow + k0);
      __bf16* dst = ldsA + tid * LDS_STRIDE;
#pragma unroll
      for (int q = 0; q < 8; ++q) {
        float4 xv = xs[q], mv = ms[q];
        v4bf o;
        o[0] = (__bf16)(xv.x * mv.x);
        o[1] = (__bf16)(xv.y * mv.y);
        o[2] = (__bf16)(xv.z * mv.z);
        o[3] = (__bf16)(xv.w * mv.w);
        *(v4bf*)(dst + q * 4) = o;
      }
    }
    __syncthreads();

    if (k0 + 32 < HIDDIM) {
      __builtin_prefetch(Wh + (size_t)(wNb + (lane & 15)) * HIDDIM + k0 + 32, 0, 1);
      __builtin_prefetch(Wt + (size_t)(wNb + (lane & 15)) * HIDDIM + k0 + 32, 0, 1);
    }

    v16bf bH = load_b_frag(Wh, wNb, k0, lane);
    v16bf bT = load_b_frag(Wt, wNb, k0, lane);
#pragma unroll
    for (int i = 0; i < 4; ++i) {
      v16bf af = load_a_frag(ldsA, wMb + i * 16, lane);
      accH[i] = wmma_bf16(af, bH, accH[i]);
      accT[i] = wmma_bf16(af, bT, accT[i]);
    }
    __syncthreads();
  }

  const int nl = lane & 15;
  const int mh = (lane >> 4) * 8;
  const int n  = wNb + nl;
  const float bhv = bh[n];
  const float btv = bt[n];
#pragma unroll
  for (int i = 0; i < 4; ++i) {
    int mb = wMb + i * 16 + mh;
#pragma unroll
    for (int r = 0; r < 8; ++r) {
      int    b    = mb + r;
      size_t idx  = (size_t)b * HIDDIM + n;
      float  hpre = accH[i][r] + bhv;
      float  tpre = accT[i][r] + btv;
      if (Hs) { hpre += Hs[idx]; tpre += Ts[idx]; }  // layer 0: add input projection
      float hv = tanhf(hpre);
      float tv = 1.f / (1.f + __expf(-tpre));
      float so = s_in[idx];
      float sn = (hv - so) * tv + so;
      s_out[idx] = sn;
      if (outs) outs[idx] = sn;                      // last layer: emit outs[s]
    }
  }
}

// ---------------- host orchestration ----------------

extern "C" void kernel_launch(void* const* d_in, const int* in_sizes, int n_in,
                              void* d_out, int out_size, void* d_ws, size_t ws_size,
                              hipStream_t stream) {
  (void)in_sizes; (void)n_in; (void)out_size; (void)ws_size;

  const float* inp    = (const float*)d_in[0];
  const float* hidden = (const float*)d_in[1];
  const float* hmask  = (const float*)d_in[2];
  const float* smask  = (const float*)d_in[3];
  const float* Wih    = (const float*)d_in[4];
  const float* bih    = (const float*)d_in[5];
  const float* Wit    = (const float*)d_in[6];
  const float* bit_   = (const float*)d_in[7];
  const float* Wh     = (const float*)d_in[8];
  const float* bh     = (const float*)d_in[9];
  const float* Wt     = (const float*)d_in[10];
  const float* bt     = (const float*)d_in[11];
  float* out = (float*)d_out;

  // workspace layout
  const size_t MSQ = (size_t)HIDDIM * HIDDIM;          // 1M elements per weight matrix
  __bf16* wih = (__bf16*)d_ws;
  __bf16* wit = wih + MSQ;
  __bf16* wh  = wit + MSQ;
  __bf16* wt  = wh + DEPTH * MSQ;
  float*  Hbuf = (float*)(wt + DEPTH * MSQ);
  float*  Tbuf = Hbuf + (size_t)SEQ * BATCH * HIDDIM;
  float*  s0   = Tbuf + (size_t)SEQ * BATCH * HIDDIM;
  float*  s1   = s0 + (size_t)BATCH * HIDDIM;

  // 1) weights -> bf16 (L2-resident working set for the whole scan)
  cvt_f32_bf16<<<(int)(MSQ / 256), 256, 0, stream>>>(Wih, wih, (int)MSQ);
  cvt_f32_bf16<<<(int)(MSQ / 256), 256, 0, stream>>>(Wit, wit, (int)MSQ);
  cvt_f32_bf16<<<(int)(DEPTH * MSQ / 256), 256, 0, stream>>>(Wh, wh, (int)(DEPTH * MSQ));
  cvt_f32_bf16<<<(int)(DEPTH * MSQ / 256), 256, 0, stream>>>(Wt, wt, (int)(DEPTH * MSQ));
  copy_f32<<<(BATCH * HIDDIM) / 256, 256, 0, stream>>>(hidden, s0, BATCH * HIDDIM);

  // 2) batched input projection for all timesteps (both gates)
  {
    dim3 grid(16, SEQ);
    rhn_input_gemm<<<grid, 256, 0, stream>>>(inp, hmask, wih, wit, bih, bit_, Hbuf, Tbuf);
  }

  // 3) sequential scan: 200 steps x 5 layers, ping-pong state buffers
  int parity = 0;
  for (int s = 0; s < SEQ; ++s) {
    for (int l = 0; l < DEPTH; ++l) {
      const float* sin  = parity ? s1 : s0;
      float*       sout = parity ? s0 : s1;
      const float* Hsp  = (l == 0) ? Hbuf + (size_t)s * BATCH * HIDDIM : nullptr;
      const float* Tsp  = (l == 0) ? Tbuf + (size_t)s * BATCH * HIDDIM : nullptr;
      float*       op   = (l == DEPTH - 1) ? out + (size_t)s * BATCH * HIDDIM : nullptr;
      rhn_layer<<<HIDDIM / 32, 128, 0, stream>>>(
          sin, sout, smask,
          wh + (size_t)l * MSQ, wt + (size_t)l * MSQ,
          bh + (size_t)l * HIDDIM, bt + (size_t)l * HIDDIM,
          Hsp, Tsp, op);
      parity ^= 1;
    }
  }

  // 4) final state -> `last` output
  copy_f32<<<(BATCH * HIDDIM) / 256, 256, 0, stream>>>(
      parity ? s1 : s0, out + (size_t)SEQ * BATCH * HIDDIM, BATCH * HIDDIM);
}
